// FlattenedVectorQuantizer_28509992911404
// MI455X (gfx1250) — compile-verified, matrix operand order checked by reading, not run on hardware
//
#include <hip/hip_runtime.h>
#include <hip/hip_bf16.h>

typedef __attribute__((ext_vector_type(16))) __bf16 v16bf;
typedef __attribute__((ext_vector_type(8)))  __bf16 v8bf;
typedef __attribute__((ext_vector_type(4)))  __bf16 v4bf;
typedef __attribute__((ext_vector_type(8)))  float  v8f;

#define N_TOKENS   16384
#define CODE_DIM   512
#define CODEBOOK   8192
#define MTILE      64      // tokens per block (4 waves x 16)
#define NTILE      32      // codes per chunk (2 WMMA N tiles per wave)
#define KBLK       64      // CODE_DIM / 8 k-blocks per row
#define COMMIT     0.25f

// ---------------------------------------------------------------------------
// Kernel 1: codebook row squared norms ||e_k||^2
// ---------------------------------------------------------------------------
__global__ __launch_bounds__(256) void vq_norms(const float* __restrict__ w,
                                                float* __restrict__ norms) {
  const int code = blockIdx.x * 8 + (threadIdx.x >> 5);
  const int lane = threadIdx.x & 31;
  const float4* row = (const float4*)(w + (size_t)code * CODE_DIM);
  float s = 0.f;
#pragma unroll
  for (int i = lane; i < CODE_DIM / 4; i += 32) {
    float4 v = row[i];
    s += v.x * v.x + v.y * v.y + v.z * v.z + v.w * v.w;
  }
#pragma unroll
  for (int off = 16; off; off >>= 1) s += __shfl_xor(s, off, 32);
  if (lane == 0) norms[code] = s;
}

// ---------------------------------------------------------------------------
// K-major tiled LDS layout: arr[group][kblock][row16][8 elems], so a bf16
// 16x32 A/B fragment is  base(row,halfsel) + kc-dependent immediate offset.
// Lanes 0-15 take K = kc+{0..7, 16..23} (kblocks kc/8, kc/8+2);
// lanes 16-31 take K = kc+{8..15, 24..31} (kblocks kc/8+1, kc/8+3).
// A wave-wide ds_load_b128 covers all 64 banks exactly twice -> no conflicts.
// ---------------------------------------------------------------------------
__device__ __forceinline__ v16bf ld_frag(const __bf16* base, int kc) {
  v8bf a = *(const v8bf*)(base + (kc >> 3) * 128);        // 128 elems / kblock
  v8bf b = *(const v8bf*)(base + (kc >> 3) * 128 + 256);  // +2 kblocks
  v16bf r;
#pragma unroll
  for (int i = 0; i < 8; ++i) { r[i] = a[i]; r[i + 8] = b[i]; }
  return r;
}

#define WMMA_BF16(A, B, C) \
  __builtin_amdgcn_wmma_f32_16x16x32_bf16(false, (A), false, (B), (short)0, (C), false, false)

// ---------------------------------------------------------------------------
// Kernel 2: fused distance GEMM (bf16x3 -> f32 WMMA) + running argmin.
// Block = 128 threads (4 waves); each wave: 16 tokens x 32 codes per chunk,
// A fragments reused across both N tiles, two independent accumulator chains.
// ---------------------------------------------------------------------------
__global__ __launch_bounds__(128) void vq_argmin(const float* __restrict__ z,
                                                 const float* __restrict__ w,
                                                 const float* __restrict__ norms,
                                                 int* __restrict__ out_idx) {
  __shared__ __bf16 zhi[4][KBLK][16][8];   // 64 KB   group = token row >> 4
  __shared__ __bf16 zlo[4][KBLK][16][8];   // 64 KB
  __shared__ __bf16 whi[2][KBLK][16][8];   // 32 KB   group = code row >> 4
  __shared__ __bf16 wlo[2][KBLK][16][8];   // 32 KB   -> 192 KB total

  const int tid     = threadIdx.x;
  const int lane    = tid & 31;
  const int wave    = tid >> 5;
  const int tokBase = blockIdx.x * MTILE;

  // --- stage the 64x512 z tile as bf16 hi/lo (error-compensated split) ---
  {
    const float4* z4 = (const float4*)(z + (size_t)tokBase * CODE_DIM);
    for (int i = tid; i < MTILE * (CODE_DIM / 4); i += 128) {
      const int row = i >> 7, c4 = i & 127;
      float4 v = z4[row * (CODE_DIM / 4) + c4];
      float vs[4] = {v.x, v.y, v.z, v.w};
      v4bf h, l;
#pragma unroll
      for (int j = 0; j < 4; ++j) {
        h[j] = (__bf16)vs[j];
        l[j] = (__bf16)(vs[j] - (float)h[j]);
      }
      __bf16* dh = &zhi[row >> 4][c4 >> 1][row & 15][(c4 & 1) * 4];
      __bf16* dl = &zlo[row >> 4][c4 >> 1][row & 15][(c4 & 1) * 4];
      *(v4bf*)dh = h;
      *(v4bf*)dl = l;
    }
  }

  const int r   = lane & 15;
  const int kb0 = (lane < 16) ? 0 : 1;           // half-wave kblock select
  const __bf16* aHi = &zhi[wave][kb0][r][0];
  const __bf16* aLo = &zlo[wave][kb0][r][0];
  const __bf16* b0H = &whi[0][kb0][r][0];
  const __bf16* b0L = &wlo[0][kb0][r][0];
  const __bf16* b1H = &whi[1][kb0][r][0];
  const __bf16* b1L = &wlo[1][kb0][r][0];

  float minv[8];
  int   mini[8];
#pragma unroll
  for (int q = 0; q < 8; ++q) { minv[q] = 3.0e38f; mini[q] = 0; }

  for (int chunk = 0; chunk < CODEBOOK / NTILE; ++chunk) {
    __syncthreads();
    // --- stage 32x512 weight chunk as bf16 hi/lo ---
    const float4* w4 = (const float4*)(w + (size_t)chunk * NTILE * CODE_DIM);
    for (int i = tid; i < NTILE * (CODE_DIM / 4); i += 128) {
      const int row = i >> 7, c4 = i & 127;
      float4 v = w4[row * (CODE_DIM / 4) + c4];
      float vs[4] = {v.x, v.y, v.z, v.w};
      v4bf h, l;
#pragma unroll
      for (int j = 0; j < 4; ++j) {
        h[j] = (__bf16)vs[j];
        l[j] = (__bf16)(vs[j] - (float)h[j]);
      }
      __bf16* dh = &whi[row >> 4][c4 >> 1][row & 15][(c4 & 1) * 4];
      __bf16* dl = &wlo[row >> 4][c4 >> 1][row & 15][(c4 & 1) * 4];
      *(v4bf*)dh = h;
      *(v4bf*)dl = l;
    }
    if (chunk + 1 < CODEBOOK / NTILE && tid < NTILE)
      __builtin_prefetch(w + ((size_t)(chunk + 1) * NTILE + tid) * CODE_DIM, 0, 0);
    __syncthreads();

    // --- two 16x16 tiles: acc_t = zl*wh_t + zh*wl_t + zh*wh_t over D=512 ---
    v8f acc0 = {};
    v8f acc1 = {};
#pragma unroll
    for (int kc = 0; kc < CODE_DIM; kc += 32) {
      v16bf ahi = ld_frag(aHi, kc);
      v16bf alo = ld_frag(aLo, kc);
      v16bf bh0 = ld_frag(b0H, kc);
      v16bf bl0 = ld_frag(b0L, kc);
      v16bf bh1 = ld_frag(b1H, kc);
      v16bf bl1 = ld_frag(b1L, kc);
      acc0 = WMMA_BF16(alo, bh0, acc0);
      acc1 = WMMA_BF16(alo, bh1, acc1);
      acc0 = WMMA_BF16(ahi, bl0, acc0);
      acc1 = WMMA_BF16(ahi, bl1, acc1);
      acc0 = WMMA_BF16(ahi, bh0, acc0);
      acc1 = WMMA_BF16(ahi, bh1, acc1);
    }

    // score = ||e||^2 - 2 z.e ; lane column n = lane&15 (+16 for tile 1),
    // row m = q + 8*(lane>=16)
    const int   c0  = chunk * NTILE + r;
    const int   c1  = c0 + 16;
    const float wn0 = norms[c0];
    const float wn1 = norms[c1];
#pragma unroll
    for (int q = 0; q < 8; ++q) {
      float s0 = wn0 - 2.0f * acc0[q];
      float s1 = wn1 - 2.0f * acc1[q];
      if (s0 < minv[q]) { minv[q] = s0; mini[q] = c0; }
      if (s1 < minv[q]) { minv[q] = s1; mini[q] = c1; }
    }
  }

  // --- argmin across the 16 lanes holding one token's columns ---
#pragma unroll
  for (int q = 0; q < 8; ++q) {
    float v = minv[q];
    int   ix = mini[q];
#pragma unroll
    for (int off = 8; off; off >>= 1) {
      float ov = __shfl_xor(v, off, 32);
      int   oi = __shfl_xor(ix, off, 32);
      if (ov < v || (ov == v && oi < ix)) { v = ov; ix = oi; }
    }
    mini[q] = ix;
  }
  if ((lane & 15) == 0) {
    const int mbase = tokBase + wave * 16 + ((lane >> 4) << 3);  // +0 or +8
#pragma unroll
    for (int q = 0; q < 8; ++q) out_idx[mbase + q] = mini[q];
  }
}

// ---------------------------------------------------------------------------
// Kernel 3: gather quantized rows, emit indices, accumulate 1.25*MSE loss.
// ---------------------------------------------------------------------------
__global__ __launch_bounds__(128) void vq_gather(const float* __restrict__ z,
                                                 const float* __restrict__ w,
                                                 const int* __restrict__ idx,
                                                 float* __restrict__ loss,
                                                 float* __restrict__ quant,
                                                 float* __restrict__ idx_out) {
  const int token = blockIdx.x;
  const int code  = idx[token];
  const float4* wr = (const float4*)(w + (size_t)code  * CODE_DIM);
  const float4* zr = (const float4*)(z + (size_t)token * CODE_DIM);
  float4*       qr = (float4*)(quant + (size_t)token * CODE_DIM);

  const int i = threadIdx.x;          // 128 threads x float4 = 512 elems
  float4 a = wr[i], b = zr[i];
  qr[i] = a;
  float dx = a.x - b.x, dy = a.y - b.y, dz = a.z - b.z, dw = a.w - b.w;
  float s = dx * dx + dy * dy + dz * dz + dw * dw;

#pragma unroll
  for (int off = 16; off; off >>= 1) s += __shfl_xor(s, off, 32);
  __shared__ float ps[4];
  if ((threadIdx.x & 31) == 0) ps[threadIdx.x >> 5] = s;
  __syncthreads();
  if (threadIdx.x == 0) {
    float t = ps[0] + ps[1] + ps[2] + ps[3];
    atomicAdd(loss, t * ((1.0f + COMMIT) / ((float)N_TOKENS * (float)CODE_DIM)));
    idx_out[token] = (float)code;
  }
}

__global__ void vq_zero(float* p) { *p = 0.0f; }

// ---------------------------------------------------------------------------
extern "C" void kernel_launch(void* const* d_in, const int* in_sizes, int n_in,
                              void* d_out, int out_size, void* d_ws, size_t ws_size,
                              hipStream_t stream) {
  const float* z = (const float*)d_in[0];   // [16384, 512]
  const float* w = (const float*)d_in[1];   // [8192, 512]

  float* out     = (float*)d_out;
  float* loss    = out;                                    // [1]
  float* quant   = out + 1;                                // [16384*512]
  float* idx_out = out + 1 + (size_t)N_TOKENS * CODE_DIM;  // [16384]

  float* norms = (float*)d_ws;                                    // 8192 f32
  int*   idx   = (int*)((char*)d_ws + CODEBOOK * sizeof(float));  // 16384 i32

  vq_zero  <<<1, 1, 0, stream>>>(loss);
  vq_norms <<<CODEBOOK / 8, 256, 0, stream>>>(w, norms);
  vq_argmin<<<N_TOKENS / MTILE, 128, 0, stream>>>(z, w, norms, idx);
  vq_gather<<<N_TOKENS, 128, 0, stream>>>(z, w, idx, loss, quant, idx_out);
}
